// HolographicThoracicMemory_50379966382326
// MI455X (gfx1250) — matrix operand order repaired and checked by reading, not compiled, not run on hardware
//
#include <hip/hip_runtime.h>
#include <hip/hip_bf16.h>

typedef __attribute__((ext_vector_type(16))) __bf16 v16bf;
typedef __attribute__((ext_vector_type(8)))  __bf16 v8bf;
typedef __attribute__((ext_vector_type(8)))  float  v8f;
typedef __attribute__((ext_vector_type(4)))  unsigned int u32x4;
typedef __attribute__((ext_vector_type(8)))  int i32x8;
typedef __attribute__((ext_vector_type(4)))  int i32x4;

#define D_SUP 1024
#define D_ANOM 128
#define D_IND 2048
#define D_IN 3200
#define H1 256
#define FEAT 128
#define DEPTH 12
#define NCOND 14
#define CAP 256
#define SIM_SCALE (0.42f * 1.8f / 256.0f)
#define ANOM_SCALE (0.3f / 128.0f)

#if __has_builtin(__builtin_amdgcn_tensor_load_to_lds) && __has_builtin(__builtin_amdgcn_s_wait_tensorcnt)
#define HAVE_TDM 1
#else
#define HAVE_TDM 0
#endif

static __device__ __forceinline__ unsigned short f2bf(float f) {
    union { float f; unsigned u; } v; v.f = f;
    unsigned r = v.u + 0x7fffu + ((v.u >> 16) & 1u);
    return (unsigned short)(r >> 16);
}
static __device__ __forceinline__ unsigned pack2(float a, float b) {
    return (unsigned)f2bf(a) | ((unsigned)f2bf(b) << 16);
}
static __device__ __forceinline__ v8f zero8() {
    v8f z;
    #pragma unroll
    for (int i = 0; i < 8; i++) z[i] = 0.f;
    return z;
}

union Frag { v16bf v; v8bf h[2]; };

// A fragment: 16x32 bf16, row-major LDS tile with element stride `S`
static __device__ __forceinline__ v16bf load_a_frag(const unsigned short* lds, int S,
                                                    int m0, int kbase, int lane) {
    int m  = m0 + (lane & 15);
    int kb = (lane >> 4) << 3;              // 0 or 8
    const unsigned short* p = lds + m * S + kbase + kb;
    Frag f;
    f.h[0] = *(const v8bf*)p;               // K = kb .. kb+7
    f.h[1] = *(const v8bf*)(p + 16);        // K = kb+16 .. kb+23
    return f.v;
}
// B fragment: 32x16 bf16, LDS holds B transposed as [N][K] with stride `S`
static __device__ __forceinline__ v16bf load_b_frag(const unsigned short* lds, int S,
                                                    int n0, int kbase, int lane) {
    int n  = n0 + (lane & 15);
    int kb = (lane >> 4) << 3;              // 0 or 8
    const unsigned short* p = lds + n * S + kbase + kb;
    Frag f;
    f.h[0] = *(const v8bf*)p;               // K = kb .. kb+7
    f.h[1] = *(const v8bf*)(p + 16);        // K = kb+16 .. kb+23
    return f.v;
}
#define WMMA_BF16(a, b, c) \
    __builtin_amdgcn_wmma_f32_16x16x32_bf16(false, (a), false, (b), (short)0, (c), false, false)

// ---- Tensor Data Mover: 2-D tile of 2-byte elements, global -> LDS, with LDS row padding ----
// D# layout per CDNA5 ISA 8.3/8.4: group0 {count, lds_addr, global_addr, type=2},
// group1 {data_size=2B, pad_enable, pad_interval, pad_amount, tensor dims, tile dims, stride}.
static __device__ __forceinline__ void tdm_load_2d_bf16(
    unsigned ldsOff, const void* gaddr, unsigned tensor_w, unsigned tensor_h,
    unsigned tile_w, unsigned tile_h, unsigned stride_elems,
    unsigned pad_interval_code, unsigned pad_amount_code) {
#if HAVE_TDM
    unsigned long long ga = (unsigned long long)(size_t)gaddr;
    u32x4 g0;
    g0[0] = 1u;                                              // count=1, no gather
    g0[1] = ldsOff;                                          // lds_addr (bytes)
    g0[2] = (unsigned)(ga & 0xffffffffu);                    // global_addr lo
    g0[3] = (unsigned)((ga >> 32) & 0x01ffffffu) | 0x80000000u; // global_addr hi | type=2
    i32x8 g1;
    g1[0] = (int)((1u << 16) |                               // data_size = 2 bytes
                  (1u << 20) |                               // pad_enable
                  (pad_interval_code << 22) |
                  (pad_amount_code << 25));
    g1[1] = (int)((tensor_w & 0xffffu) << 16);               // tensor_dim0 lo16
    g1[2] = (int)((tensor_w >> 16) | ((tensor_h & 0xffffu) << 16)); // td0 hi | td1 lo
    g1[3] = (int)(((tensor_h >> 16) & 0xffffu) | (tile_w << 16));   // td1 hi | tile_dim0
    g1[4] = (int)(tile_h & 0xffffu);                         // tile_dim1 | tile_dim2=0
    g1[5] = (int)stride_elems;                               // tensor_dim0_stride lo32
    g1[6] = 0;
    g1[7] = 0;
    i32x4 gz;
    gz[0] = 0; gz[1] = 0; gz[2] = 0; gz[3] = 0;
    i32x8 gz8;
    #pragma unroll
    for (int i = 0; i < 8; i++) gz8[i] = 0;
    __builtin_amdgcn_tensor_load_to_lds(g0, g1, gz, gz, gz8, 0);
#endif
}
static __device__ __forceinline__ void tdm_wait() {
#if HAVE_TDM
    __builtin_amdgcn_s_wait_tensorcnt(0);                    // no-op for non-issuing waves
#endif
}

// ====================== Kernel 1: h1 = relu(concat(X) @ W1 + b1), bf16 out ======================
__global__ __launch_bounds__(256) void k_gemm1(
    const float* __restrict__ sup, const float* __restrict__ anom, const float* __restrict__ ind,
    const float* __restrict__ W1, const float* __restrict__ bias1,
    unsigned short* __restrict__ h1) {
    __shared__ __align__(16) unsigned short lA[128 * 40];
    __shared__ __align__(16) unsigned short lB[128 * 40];
    const int t    = threadIdx.x;
    const int lane = t & 31;
    const int w    = t >> 5;
    const int m0   = (w >> 2) * 64;
    const int n0   = (w & 3) * 32;
    const int bM   = blockIdx.x;
    const int nB   = blockIdx.y;

    v8f acc[4][2];
    #pragma unroll
    for (int i = 0; i < 4; i++)
        #pragma unroll
        for (int j = 0; j < 2; j++) acc[i][j] = zero8();

    const int r0 = t >> 3;
    const int cq = (t & 7) * 4;

    for (int k0 = 0; k0 < D_IN; k0 += 32) {
        const float* src; int scol, sw;
        if (k0 < D_SUP)               { src = sup;  scol = k0;                    sw = D_SUP;  }
        else if (k0 < D_SUP + D_ANOM) { src = anom; scol = k0 - D_SUP;            sw = D_ANOM; }
        else                          { src = ind;  scol = k0 - (D_SUP + D_ANOM); sw = D_IND;  }
        #pragma unroll
        for (int i = 0; i < 4; i++) {
            int row = r0 + 32 * i;
            const float4 x = *(const float4*)(src + (size_t)(bM * 128 + row) * sw + scol + cq);
            unsigned* dst = (unsigned*)&lA[row * 40 + cq];
            dst[0] = pack2(x.x, x.y);
            dst[1] = pack2(x.z, x.w);
            __builtin_prefetch(src + (size_t)(bM * 128 + row) * sw + scol + cq + 32, 0, 0);
        }
        {
            int kk = t >> 3;
            #pragma unroll
            for (int i = 0; i < 4; i++) {
                int c = cq + 32 * i;
                const float4 ww = *(const float4*)(W1 + (size_t)(k0 + kk) * H1 + nB * 128 + c);
                lB[(c + 0) * 40 + kk] = f2bf(ww.x);
                lB[(c + 1) * 40 + kk] = f2bf(ww.y);
                lB[(c + 2) * 40 + kk] = f2bf(ww.z);
                lB[(c + 3) * 40 + kk] = f2bf(ww.w);
            }
        }
        __syncthreads();
        v16bf aF[4];
        #pragma unroll
        for (int mt = 0; mt < 4; mt++) aF[mt] = load_a_frag(lA, 40, m0 + mt * 16, 0, lane);
        #pragma unroll
        for (int nt = 0; nt < 2; nt++) {
            v16bf bF = load_b_frag(lB, 40, n0 + nt * 16, 0, lane);
            #pragma unroll
            for (int mt = 0; mt < 4; mt++) acc[mt][nt] = WMMA_BF16(aF[mt], bF, acc[mt][nt]);
        }
        __syncthreads();
    }
    #pragma unroll
    for (int nt = 0; nt < 2; nt++) {
        int col = nB * 128 + n0 + nt * 16 + (lane & 15);
        float bv = bias1[col];
        #pragma unroll
        for (int mt = 0; mt < 4; mt++) {
            #pragma unroll
            for (int r = 0; r < 8; r++) {
                int row = bM * 128 + m0 + mt * 16 + r + ((lane >> 4) << 3);
                float v = acc[mt][nt][r] + bv;
                v = v > 0.f ? v : 0.f;
                h1[(size_t)row * H1 + col] = f2bf(v);
            }
        }
    }
}

// ============ Kernel 2a: memory_state = 12 layers(relu(h1 @ W2 + b2)), + anomaly weights ============
__global__ __launch_bounds__(256) void k_mem(
    const unsigned short* __restrict__ h1, const float* __restrict__ anomaly,
    const float* __restrict__ W2, const float* __restrict__ b2,
    const float* __restrict__ consW, const float* __restrict__ consB,
    unsigned short* __restrict__ memg, float* __restrict__ awg) {
    __shared__ __align__(16) unsigned short lM[128 * 136];
    __shared__ __align__(16) unsigned short lX[128 * 136];
    __shared__ __align__(16) unsigned short lY[128 * 40];
    __shared__ float anomW[128];
    const int t    = threadIdx.x;
    const int lane = t & 31;
    const int w    = t >> 5;
    const int m0   = w * 16;
    const int bRow = blockIdx.x * 128;

    if (t < 128) anomW[t] = 0.f;
    __syncthreads();
    {
        int row = t >> 1, off = (t & 1) * 64;
        const float* p = anomaly + (size_t)(bRow + row) * D_ANOM + off;
        float s = 0.f;
        #pragma unroll
        for (int j = 0; j < 64; j++) s += fabsf(p[j]);
        atomicAdd(&anomW[row], s);
    }

    // ---- GEMM2: relu(h1 @ W2 + b2), K=256, N=128 ----
    v8f acc[8];
    #pragma unroll
    for (int nt = 0; nt < 8; nt++) acc[nt] = zero8();
    for (int kc = 0; kc < 8; kc++) {
        __syncthreads();
#if HAVE_TDM
        if (w == 0)  // wave-uniform descriptor; TDM tile = [128 rows][32 bf16], LDS rows padded to 40
            tdm_load_2d_bf16((unsigned)(size_t)&lY[0], h1 + (size_t)bRow * H1 + kc * 32,
                             H1, 16384u, 32u, 128u, H1, /*16 DW*/3u, /*4 DW*/3u);
#else
        {
            int r0 = t >> 3, cq = (t & 7) * 4;
            #pragma unroll
            for (int i = 0; i < 4; i++) {
                int row = r0 + 32 * i;
                *(uint2*)&lY[row * 40 + cq] =
                    *(const uint2*)(h1 + (size_t)(bRow + row) * H1 + kc * 32 + cq);
            }
        }
#endif
        {   // W2 chunk transposed: W2[kc*32+kk][c] -> lX[c][kk]
            int kk = t >> 3, cq = (t & 7) * 4;
            #pragma unroll
            for (int i = 0; i < 4; i++) {
                int c = cq + 32 * i;
                const float4 ww = *(const float4*)(W2 + (size_t)(kc * 32 + kk) * FEAT + c);
                lX[(c + 0) * 40 + kk] = f2bf(ww.x);
                lX[(c + 1) * 40 + kk] = f2bf(ww.y);
                lX[(c + 2) * 40 + kk] = f2bf(ww.z);
                lX[(c + 3) * 40 + kk] = f2bf(ww.w);
            }
        }
        tdm_wait();
        __syncthreads();
        v16bf aF = load_a_frag(lY, 40, m0, 0, lane);
        #pragma unroll
        for (int nt = 0; nt < 8; nt++) {
            v16bf bF = load_b_frag(lX, 40, nt * 16, 0, lane);
            acc[nt] = WMMA_BF16(aF, bF, acc[nt]);
        }
    }
    #pragma unroll
    for (int nt = 0; nt < 8; nt++) {
        int col = nt * 16 + (lane & 15);
        float bv = b2[col];
        #pragma unroll
        for (int r = 0; r < 8; r++) {
            int row = m0 + r + ((lane >> 4) << 3);
            float v = acc[nt][r] + bv;
            v = v > 0.f ? v : 0.f;
            lM[row * 136 + col] = f2bf(v);
        }
    }

    // ---- 12 consolidator layers ----
    for (int l = 0; l < DEPTH; l++) {
        __syncthreads();
        {
            int kk0 = t >> 5, cq = (t & 31) * 4;
            #pragma unroll
            for (int ki = 0; ki < 16; ki++) {
                int kk = kk0 + 8 * ki;
                const float4 ww = *(const float4*)(consW + (size_t)l * FEAT * FEAT + (size_t)kk * FEAT + cq);
                lX[(cq + 0) * 136 + kk] = f2bf(ww.x);
                lX[(cq + 1) * 136 + kk] = f2bf(ww.y);
                lX[(cq + 2) * 136 + kk] = f2bf(ww.z);
                lX[(cq + 3) * 136 + kk] = f2bf(ww.w);
            }
        }
        __syncthreads();
        v8f a2[8];
        #pragma unroll
        for (int nt = 0; nt < 8; nt++) a2[nt] = zero8();
        #pragma unroll
        for (int kc = 0; kc < 4; kc++) {
            v16bf aF = load_a_frag(lM, 136, m0, kc * 32, lane);
            #pragma unroll
            for (int nt = 0; nt < 8; nt++) {
                v16bf bF = load_b_frag(lX, 136, nt * 16, kc * 32, lane);
                a2[nt] = WMMA_BF16(aF, bF, a2[nt]);
            }
        }
        #pragma unroll
        for (int nt = 0; nt < 8; nt++) {
            int col = nt * 16 + (lane & 15);
            float bv = consB[l * FEAT + col];
            #pragma unroll
            for (int r = 0; r < 8; r++) {
                int row = m0 + r + ((lane >> 4) << 3);
                float v = a2[nt][r] + bv;
                v = v > 0.f ? v : 0.f;
                lM[row * 136 + col] = f2bf(v);
            }
        }
    }
    __syncthreads();
    // write memory_state tile (bf16) + scaled anomaly weights
    {
        int row = t >> 1, part = (t & 1) * 64;
        #pragma unroll
        for (int i = 0; i < 8; i++) {
            *(uint4*)(memg + (size_t)(bRow + row) * FEAT + part + i * 8) =
                *(const uint4*)&lM[row * 136 + part + i * 8];
        }
        if (t < 128) awg[bRow + t] = anomW[t] * ANOM_SCALE;
    }
}

// ============ Kernel 2b: one (row-tile, bank) pair per block — sims + retrieval + output ============
__global__ __launch_bounds__(256) void k_banks(
    const unsigned short* __restrict__ memg, const float* __restrict__ awg,
    const float* __restrict__ refP, const float* __restrict__ objP,
    const float* __restrict__ retW, const float* __restrict__ retB,
    float* __restrict__ out) {
    __shared__ __align__(16) unsigned short lM[128 * 136];
    __shared__ __align__(16) unsigned short lX[128 * 136];
    __shared__ __align__(16) unsigned short lY[128 * 136];
    __shared__ float rowsum[128];
    __shared__ float law[128];
    const int t    = threadIdx.x;
    const int lane = t & 31;
    const int w    = t >> 5;
    const int m0   = w * 16;
    const int bRow = blockIdx.x * 128;
    const int k    = blockIdx.y;           // bank

    if (t < 128) { rowsum[t] = 0.f; law[t] = awg[bRow + t]; }
#if HAVE_TDM
    if (w == 0)  // memory_state tile [128][128] bf16, LDS rows padded to 136
        tdm_load_2d_bf16((unsigned)(size_t)&lM[0], memg + (size_t)bRow * FEAT,
                         FEAT, 16384u, FEAT, 128u, FEAT, /*32 DW*/4u, /*4 DW*/3u);
    tdm_wait();
#else
    {
        int row = t >> 1, part = (t & 1) * 64;
        #pragma unroll
        for (int i = 0; i < 8; i++) {
            *(uint4*)&lM[row * 136 + part + i * 8] =
                *(const uint4*)(memg + (size_t)(bRow + row) * FEAT + part + i * 8);
        }
    }
#endif
    __syncthreads();

    v16bf mF[4];
    #pragma unroll
    for (int kc = 0; kc < 4; kc++) mF[kc] = load_a_frag(lM, 136, m0, kc * 32, lane);

    float partial[8];
    #pragma unroll
    for (int r = 0; r < 8; r++) partial[r] = 0.f;

    for (int hh = 0; hh < 2; hh++) {       // CAP=256 in two halves of 128
        __syncthreads();
        {   // ref/obj are [C][D] = [N][K]: direct convert copy
            int cc0 = t >> 5, cq = (t & 31) * 4;
            #pragma unroll
            for (int ci = 0; ci < 16; ci++) {
                int cc = cc0 + 8 * ci;
                size_t go = ((size_t)k * CAP + hh * 128 + cc) * FEAT + cq;
                const float4 rv = *(const float4*)(refP + go);
                unsigned* d1 = (unsigned*)&lX[cc * 136 + cq];
                d1[0] = pack2(rv.x, rv.y); d1[1] = pack2(rv.z, rv.w);
                const float4 ov = *(const float4*)(objP + go);
                unsigned* d2 = (unsigned*)&lY[cc * 136 + cq];
                d2[0] = pack2(ov.x, ov.y); d2[1] = pack2(ov.z, ov.w);
            }
        }
        __syncthreads();
        #pragma unroll
        for (int cch = 0; cch < 8; cch++) {
            v8f rA = zero8(), oA = zero8();
            #pragma unroll
            for (int kc = 0; kc < 4; kc++) {
                v16bf bF = load_b_frag(lX, 136, cch * 16, kc * 32, lane);
                rA = WMMA_BF16(mF[kc], bF, rA);
            }
            #pragma unroll
            for (int kc = 0; kc < 4; kc++) {
                v16bf bF = load_b_frag(lY, 136, cch * 16, kc * 32, lane);
                oA = WMMA_BF16(mF[kc], bF, oA);
            }
            #pragma unroll
            for (int r = 0; r < 8; r++) partial[r] += rA[r] * oA[r];
        }
    }
    #pragma unroll
    for (int r = 0; r < 8; r++) {
        int m = m0 + r + ((lane >> 4) << 3);
        atomicAdd(&rowsum[m], partial[r]);
    }
    __syncthreads();
    {   // retW[k][d][e] -> lX[e][d]
        int kk0 = t >> 5, cq = (t & 31) * 4;
        #pragma unroll
        for (int ki = 0; ki < 16; ki++) {
            int kk = kk0 + 8 * ki;
            const float4 ww = *(const float4*)(retW + (size_t)k * FEAT * FEAT + (size_t)kk * FEAT + cq);
            lX[(cq + 0) * 136 + kk] = f2bf(ww.x);
            lX[(cq + 1) * 136 + kk] = f2bf(ww.y);
            lX[(cq + 2) * 136 + kk] = f2bf(ww.z);
            lX[(cq + 3) * 136 + kk] = f2bf(ww.w);
        }
    }
    __syncthreads();
    v8f rc[8];
    #pragma unroll
    for (int nt = 0; nt < 8; nt++) rc[nt] = zero8();
    #pragma unroll
    for (int kc = 0; kc < 4; kc++) {
        #pragma unroll
        for (int nt = 0; nt < 8; nt++) {
            v16bf bF = load_b_frag(lX, 136, nt * 16, kc * 32, lane);
            rc[nt] = WMMA_BF16(mF[kc], bF, rc[nt]);
        }
    }
    #pragma unroll
    for (int nt = 0; nt < 8; nt++) {
        int col = nt * 16 + (lane & 15);
        float bb = retB[k * FEAT + col];
        #pragma unroll
        for (int r = 0; r < 8; r++) {
            int m = m0 + r + ((lane >> 4) << 3);
            float cw = rowsum[m] * SIM_SCALE + law[m];
            out[((size_t)(bRow + m) * NCOND + k) * FEAT + col] = (rc[nt][r] + bb) * cw;
        }
    }
}

extern "C" void kernel_launch(void* const* d_in, const int* in_sizes, int n_in,
                              void* d_out, int out_size, void* d_ws, size_t ws_size,
                              hipStream_t stream) {
    const float* sup  = (const float*)d_in[0];
    const float* anom = (const float*)d_in[1];
    const float* ind  = (const float*)d_in[2];
    const float* W1   = (const float*)d_in[3];
    const float* b1   = (const float*)d_in[4];
    const float* W2   = (const float*)d_in[5];
    const float* b2   = (const float*)d_in[6];
    const float* cW   = (const float*)d_in[7];
    const float* cB   = (const float*)d_in[8];
    const float* rP   = (const float*)d_in[9];
    const float* oP   = (const float*)d_in[10];
    const float* rW   = (const float*)d_in[11];
    const float* rB   = (const float*)d_in[12];

    char* ws = (char*)d_ws;
    unsigned short* h1   = (unsigned short*)ws;                       // 16384x256 bf16 = 8 MB
    unsigned short* memg = (unsigned short*)(ws + (size_t)16384 * H1 * 2);   // 16384x128 bf16 = 4 MB
    float*          awg  = (float*)(ws + (size_t)16384 * H1 * 2 + (size_t)16384 * FEAT * 2); // 64 KB

    dim3 g1(128, 2);
    k_gemm1<<<g1, 256, 0, stream>>>(sup, anom, ind, W1, b1, h1);
    k_mem<<<128, 256, 0, stream>>>(h1, anom, W2, b2, cW, cB, memg, awg);
    dim3 g3(128, NCOND);
    k_banks<<<g3, 256, 0, stream>>>(memg, awg, rP, oP, rW, rB, (float*)d_out);
}
